// SparseAttention_88579405512809
// MI455X (gfx1250) — matrix operand order repaired and checked by reading, not compiled, not run on hardware
//
#include <hip/hip_runtime.h>
#include <hip/hip_bf16.h>

typedef __attribute__((ext_vector_type(16))) __bf16 v16bf;
typedef __attribute__((ext_vector_type(8)))  __bf16 v8bf;
typedef __attribute__((ext_vector_type(8)))  float  v8f;
typedef __attribute__((ext_vector_type(4)))  unsigned int u32x4;
typedef __attribute__((ext_vector_type(8)))  int i32x8;
typedef __attribute__((ext_vector_type(4)))  int i32x4;

#define WMMA_BF16(A, B, C) \
    __builtin_amdgcn_wmma_f32_16x16x32_bf16(false, (A), false, (B), (short)0, (C), false, false)

// ---------------------------------------------------------------------------
// Problem constants: N=131072, e=512, h=8, s=64
// ---------------------------------------------------------------------------
#define E_DIM   512
#define S_HEAD  64
#define ROWS_PER_BLOCK 64          // 4 subtiles of 16 rows
#define NSUB    4

#define QKV_ELEMS (3 * 8 * 8 * 512)     // 98304 bf16
#define WU_ELEMS  (32 * 16 * 512)       // 262144 bf16

// x staging: 16 rows x 512 f32 per subtile, TDM pads 4 dwords every 256 dwords
// -> 8192 + 32*4 = 8320 dwords = 33280 B per buffer
#define XS_DWORDS 8320

__device__ __forceinline__ unsigned short f32_to_bf16_rne(float f) {
    unsigned int u = __float_as_uint(f);
    unsigned int r = u + 0x7FFFu + ((u >> 16) & 1u);
    return (unsigned short)(r >> 16);
}

// LDS byte offset of dword index i inside a TDM-padded x buffer
__device__ __forceinline__ int xs_off(int i) { return (i << 2) + ((i >> 8) << 4); }

// XOR-swizzled (16B granule) index into the 64x512 bf16 attn tile
__device__ __forceinline__ int attn_idx(int row, int col) {
    int chunk  = col >> 3;          // 0..63 (8 bf16 = 16B per chunk)
    int within = col & 7;
    return row * E_DIM + (((chunk ^ (row & 15)) << 3) | within);
}

// ---------------------------------------------------------------------------
// TDM: DMA one 16x512 fp32 tile (row-major, contiguous) into LDS with padding.
// D# field packing per CDNA5 ISA; 6-arg builtin form.
// ---------------------------------------------------------------------------
__device__ __forceinline__ void tdm_load_x_tile(unsigned lds_off, const float* gptr) {
    unsigned long long ga = (unsigned long long)gptr;
    u32x4 g0;
    g0[0] = 1u;                                            // count=1, user mode
    g0[1] = lds_off;                                       // lds_addr (bytes)
    g0[2] = (unsigned)(ga & 0xFFFFFFFFull);                // global_addr[31:0]
    g0[3] = (unsigned)((ga >> 32) & 0x01FFFFFFull)         // global_addr[56:32]
          | (2u << 30);                                    // type = 2 ("image")
    i32x8 g1;
    g1[0] = (int)((2u << 16)      // data_size = 2 -> 4 bytes
                | (1u << 20)      // pad_enable
                | (7u << 22)      // pad_interval code 7 -> every 256 dwords
                | (3u << 25));    // pad_amount code 3 -> 4 dwords
    g1[1] = (int)(512u << 16);    // tensor_dim0[15:0] = 512 (bits 79:48)
    g1[2] = (int)(16u  << 16);    // tensor_dim1[15:0] = 16  (bits 111:80)
    g1[3] = (int)(512u << 16);    // tile_dim0 = 512         (bits 127:112)
    g1[4] = 16;                   // tile_dim1 = 16, tile_dim2 = 0
    g1[5] = 512;                  // tensor_dim0_stride[31:0] = 512 elems
    g1[6] = 0;                    // stride0 hi / dim1_stride lo
    g1[7] = 0;
    i32x4 z4 = (i32x4)0;          // groups 2/3 unused (2D tensor)
    i32x8 z8 = (i32x8)0;          // extra descriptor group (unused)
    __builtin_amdgcn_tensor_load_to_lds(g0, g1, z4, z4, z8, 0);
}

// ---------------------------------------------------------------------------
// Prepack fp32 weights -> bf16 WMMA B-fragment layout (lane L: N=L&15,
// half j: K = kstep*32 + 16*(j>=8) + 8*(L>=16) + (j&7)).  One v16bf (32B) per lane per frag.
// ---------------------------------------------------------------------------
__global__ __launch_bounds__(256) void prepack_kernel(
    const float* __restrict__ Wq, const float* __restrict__ Wk,
    const float* __restrict__ Wv, const float* __restrict__ Wu,
    unsigned short* __restrict__ wsq, unsigned short* __restrict__ wsu)
{
    const int idx = blockIdx.x * 256 + threadIdx.x;

    if (idx < QKV_ELEMS) {
        // [proj(3)][head(8)][frag(8)=nt*2+ks][elem(512)=lane*16+j]
        int p    = idx / 32768;
        int rem  = idx % 32768;
        int head = rem / 4096;
        int rem2 = rem % 4096;
        int frag = rem2 / 512;
        int e    = rem2 % 512;
        int nt   = frag >> 1;
        int ks   = frag & 1;
        int lane = e >> 4;
        int j    = e & 15;
        int k    = ks * 32 + ((j >> 3) << 4) + ((lane >= 16) ? 8 : 0) + (j & 7);
        int col  = nt * 16 + (lane & 15);
        const float* W = (p == 0) ? Wq : ((p == 1) ? Wk : Wv);
        wsq[idx] = f32_to_bf16_rne(W[head * (64 * 64) + k * 64 + col]);
    }

    if (idx < WU_ELEMS) {
        // [ntileU(32)][kstepU(16)][elem(512)=lane*16+j]
        int nt   = idx / 8192;
        int ks   = (idx / 512) % 16;
        int e    = idx % 512;
        int lane = e >> 4;
        int j    = e & 15;
        int k    = ks * 32 + ((j >> 3) << 4) + ((lane >= 16) ? 8 : 0) + (j & 7);
        int col  = nt * 16 + (lane & 15);
        wsu[idx] = f32_to_bf16_rne(Wu[k * E_DIM + col]);
    }
}

// ---------------------------------------------------------------------------
// Fused kernel.  Block = 256 threads (8 waves), 64 rows.
//   Stage 1: wave w = head w; x subtiles arrive via TDM double-buffer.
//            Phased to cap live registers: (q,k) -> softmax -> (v).
//   Stage 2: wave w = output columns [64w, 64w+64).
// ---------------------------------------------------------------------------
__global__ __launch_bounds__(256) void fused_attention_kernel(
    const float* __restrict__ x,
    const float* __restrict__ bq, const float* __restrict__ bk,
    const float* __restrict__ bv, const float* __restrict__ bu,
    const __bf16* __restrict__ Bqkv,
    const __bf16* __restrict__ Bu,
    float* __restrict__ out)
{
    __shared__ float  xs[2][XS_DWORDS];               // 66,560 B (TDM dest)
    __shared__ __bf16 attn[ROWS_PER_BLOCK * E_DIM];   // 65,536 B, XOR-swizzled

    const int tid   = threadIdx.x;
    const int wave  = tid >> 5;
    const int lane  = tid & 31;
    const int lhalf = lane >> 4;
    const int l15   = lane & 15;
    const long rowbase = (long)blockIdx.x * ROWS_PER_BLOCK;
    const float scale = 0.044194173824159216f;   // 1/sqrt(512)

    // ---- kick off first x subtile DMA ----
    if (wave == 0) {
        unsigned lds0 = (unsigned)(unsigned long long)(const void*)&xs[0][0];
        tdm_load_x_tile(lds0, x + rowbase * E_DIM);
    }

    // ================= Stage 1: head = wave ==================
    const int hh = wave;

#pragma unroll 1
    for (int sub = 0; sub < NSUB; ++sub) {
        if (wave == 0) {
            __builtin_amdgcn_s_wait_tensorcnt(0);   // current buffer landed
        }
        __syncthreads();                             // publish to all waves
        if (wave == 0 && sub + 1 < NSUB) {           // prefetch next subtile
            unsigned ldsn = (unsigned)(unsigned long long)(const void*)&xs[(sub + 1) & 1][0];
            tdm_load_x_tile(ldsn, x + (rowbase + (sub + 1) * 16) * E_DIM);
        }

        const char* xsbuf = (const char*)&xs[sub & 1][0];
        const int   koffA = lhalf * 8;

        // ---- A fragments from LDS x (fp32 -> bf16), ks = 0,1 ----
        v16bf A[2];
#pragma unroll
        for (int ks = 0; ks < 2; ++ks) {
            const int cb = hh * S_HEAD + ks * 32;
            const int i0 = l15 * E_DIM + cb + koffA;
            const int i2 = l15 * E_DIM + cb + 16 + koffA;
            float4 f0 = *(const float4*)(xsbuf + xs_off(i0));
            float4 f1 = *(const float4*)(xsbuf + xs_off(i0 + 4));
            float4 f2 = *(const float4*)(xsbuf + xs_off(i2));
            float4 f3 = *(const float4*)(xsbuf + xs_off(i2 + 4));
            v16bf a;
            a[0]  = (__bf16)f0.x; a[1]  = (__bf16)f0.y; a[2]  = (__bf16)f0.z; a[3]  = (__bf16)f0.w;
            a[4]  = (__bf16)f1.x; a[5]  = (__bf16)f1.y; a[6]  = (__bf16)f1.z; a[7]  = (__bf16)f1.w;
            a[8]  = (__bf16)f2.x; a[9]  = (__bf16)f2.y; a[10] = (__bf16)f2.z; a[11] = (__bf16)f2.w;
            a[12] = (__bf16)f3.x; a[13] = (__bf16)f3.y; a[14] = (__bf16)f3.z; a[15] = (__bf16)f3.w;
            A[ks] = a;
        }

        // ---- phase 1: q, k via WMMA ----
        v8f q[4], k[4];
#pragma unroll
        for (int nt = 0; nt < 4; ++nt) { q[nt] = (v8f)0.0f; k[nt] = (v8f)0.0f; }
#pragma unroll
        for (int ks = 0; ks < 2; ++ks) {
#pragma unroll
            for (int nt = 0; nt < 4; ++nt) {
                const int frag = nt * 2 + ks;
                const v16bf bqf = *(const v16bf*)(Bqkv + (((0 * 8 + hh) * 8 + frag) * 512) + lane * 16);
                q[nt] = WMMA_BF16(A[ks], bqf, q[nt]);
                const v16bf bkf = *(const v16bf*)(Bqkv + (((1 * 8 + hh) * 8 + frag) * 512) + lane * 16);
                k[nt] = WMMA_BF16(A[ks], bkf, k[nt]);
            }
        }

        // ---- phase 2: softmax over head axis; q,k die into ev ----
        // C layout: lane L, reg r -> M = r + 8*lhalf, N = l15 + 16*nt.
        float ev[4][8];
        float rs[8];
#pragma unroll
        for (int r = 0; r < 8; ++r) {
            float t = 0.0f;
#pragma unroll
            for (int nt = 0; nt < 4; ++nt) {
                int   col = hh * S_HEAD + nt * 16 + l15;
                float qq  = q[nt][r] + bq[col];
                float kk  = k[nt][r] + bk[col];
                float e   = __expf(qq * kk * scale);
                ev[nt][r] = e;
                t += e;
            }
            rs[r] = t;
        }
#pragma unroll
        for (int m = 1; m <= 8; m <<= 1) {
#pragma unroll
            for (int r = 0; r < 8; ++r) rs[r] += __shfl_xor(rs[r], m, 32);
        }
#pragma unroll
        for (int r = 0; r < 8; ++r) {
            const float inv = 1.0f / rs[r];
#pragma unroll
            for (int nt = 0; nt < 4; ++nt) ev[nt][r] *= inv;
        }

        // ---- phase 3: v via WMMA, then attn = a * (v + bv) -> LDS ----
        v8f v[4];
#pragma unroll
        for (int nt = 0; nt < 4; ++nt) v[nt] = (v8f)0.0f;
#pragma unroll
        for (int ks = 0; ks < 2; ++ks) {
#pragma unroll
            for (int nt = 0; nt < 4; ++nt) {
                const int frag = nt * 2 + ks;
                const v16bf bvf = *(const v16bf*)(Bqkv + (((2 * 8 + hh) * 8 + frag) * 512) + lane * 16);
                v[nt] = WMMA_BF16(A[ks], bvf, v[nt]);
            }
        }

#pragma unroll
        for (int r = 0; r < 8; ++r) {
            const int arow = sub * 16 + r + lhalf * 8;
#pragma unroll
            for (int nt = 0; nt < 4; ++nt) {
                const int col = hh * S_HEAD + nt * 16 + l15;
                float o = ev[nt][r] * (v[nt][r] + bv[col]);
                attn[attn_idx(arow, col)] = (__bf16)o;
            }
        }
    }

    __syncthreads();

    // ========== Stage 2: attn[64,512] @ Wu -> cols [64*wave, +64) ==========
    {
        v8f acc[NSUB][4];
#pragma unroll
        for (int sub = 0; sub < NSUB; ++sub)
#pragma unroll
            for (int nt = 0; nt < 4; ++nt) acc[sub][nt] = (v8f)0.0f;

#pragma unroll 2
        for (int ks = 0; ks < 16; ++ks) {
            // A fragments for all 4 row-subtiles (swizzled 16B chunks)
            v16bf Aa[NSUB];
            const int c0 = ks * 4 + lhalf;     // chunk of halves 0..7
#pragma unroll
            for (int sub = 0; sub < NSUB; ++sub) {
                const int row = sub * 16 + l15;
                const __bf16* rbase = attn + row * E_DIM;
                v8bf lo = *(const v8bf*)(rbase + (((c0)     ^ (row & 15)) << 3));
                v8bf hi = *(const v8bf*)(rbase + (((c0 + 2) ^ (row & 15)) << 3));
                Aa[sub] = __builtin_shufflevector(lo, hi,
                                                  0, 1, 2, 3, 4, 5, 6, 7,
                                                  8, 9, 10, 11, 12, 13, 14, 15);
            }

#pragma unroll
            for (int nt = 0; nt < 4; ++nt) {
                const __bf16* bp = Bu + (((wave * 4 + nt) * 16 + ks) * 512) + lane * 16;
                if (ks + 1 < 16) __builtin_prefetch(bp + 512, 0, 1);
                const v16bf b = *(const v16bf*)bp;
#pragma unroll
                for (int sub = 0; sub < NSUB; ++sub)
                    acc[sub][nt] = WMMA_BF16(Aa[sub], b, acc[sub][nt]);
            }
        }

        // ---- epilogue: + bu, fp32 store ----
#pragma unroll
        for (int nt = 0; nt < 4; ++nt) {
            const int col  = wave * S_HEAD + nt * 16 + l15;
            const float bb = bu[col];
#pragma unroll
            for (int sub = 0; sub < NSUB; ++sub) {
#pragma unroll
                for (int r = 0; r < 8; ++r) {
                    const long row = rowbase + sub * 16 + r + lhalf * 8;
                    out[row * E_DIM + col] = acc[sub][nt][r] + bb;
                }
            }
        }
    }
}

// ---------------------------------------------------------------------------
extern "C" void kernel_launch(void* const* d_in, const int* in_sizes, int n_in,
                              void* d_out, int out_size, void* d_ws, size_t ws_size,
                              hipStream_t stream) {
    const float* x  = (const float*)d_in[0];
    const float* Wq = (const float*)d_in[1];
    const float* bq = (const float*)d_in[2];
    const float* Wk = (const float*)d_in[3];
    const float* bk = (const float*)d_in[4];
    const float* Wv = (const float*)d_in[5];
    const float* bv = (const float*)d_in[6];
    const float* Wu = (const float*)d_in[7];
    const float* bu = (const float*)d_in[8];

    unsigned short* wsq = (unsigned short*)d_ws;
    unsigned short* wsu = wsq + QKV_ELEMS;

    prepack_kernel<<<(WU_ELEMS + 255) / 256, 256, 0, stream>>>(Wq, Wk, Wv, Wu, wsq, wsu);

    const int N = in_sizes[0] / E_DIM;   // 131072
    fused_attention_kernel<<<N / ROWS_PER_BLOCK, 256, 0, stream>>>(
        x, bq, bk, bv, bu,
        (const __bf16*)wsq, (const __bf16*)wsu,
        (float*)d_out);
}
